// GNU_Model_58884001628361
// MI455X (gfx1250) — matrix operand order repaired and checked by reading, not compiled, not run on hardware
//
#include <hip/hip_runtime.h>
#include <stdint.h>

#define NN     4096      // nodes
#define VD     8192      // v dim
#define NE     65536     // raw edges
#define NEDGE  69632     // edges + self loops
#define NSTEPS 100
#define GNN_WGS 16       // workgroups participating in the GNN global barrier
#define KSPLIT 8         // K partitions in the WMMA GEMV (4096 waves total)

typedef __attribute__((ext_vector_type(16))) __bf16   v16bf;
typedef __attribute__((ext_vector_type(8)))  float    v8f;
typedef __attribute__((ext_vector_type(4)))  unsigned v4u;

union BfPack { struct { v4u lo; v4u hi; } s; v16bf v; };

__device__ __forceinline__ uint16_t f2bf(float x) {  // fp32 -> bf16, RNE
  unsigned u = __float_as_uint(x);
  u += 0x7FFFu + ((u >> 16) & 1u);
  return (uint16_t)(u >> 16);
}

// ---------------- device-wide split barrier over GNN_WGS workgroups ----------------
__device__ __forceinline__ void gbar(unsigned* bar) {
  __syncthreads();
  __threadfence();                       // release our stores to agent scope
  if (threadIdx.x == 0) {
    unsigned g = atomicAdd(&bar[1], 0u); // current generation
    if (atomicAdd(&bar[0], 1u) == GNN_WGS - 1u) {
      atomicExch(&bar[0], 0u);
      atomicAdd(&bar[1], 1u);            // release generation
    } else {
      while (atomicAdd(&bar[1], 0u) == g) __builtin_amdgcn_s_sleep(2);
    }
  }
  __syncthreads();
  __threadfence();                       // acquire others' stores
}

// ---------------- one-time precompute kernels ----------------
__global__ void k_init(float* va, float* rs, unsigned* deg, unsigned* bar) {
  int i = blockIdx.x * blockDim.x + threadIdx.x;
  if (i < VD) va[i] = 0.0f;              // v0 = zeros
  if (i < NN) { rs[i] = 0.0f; deg[i] = 1u; }  // deg starts at 1 (self loop)
  if (i < 2)  bar[i] = 0u;
}

__global__ void k_count(const int* __restrict__ ei, unsigned* __restrict__ deg) {
  int e = blockIdx.x * blockDim.x + threadIdx.x;
  if (e < NE) atomicAdd(&deg[ei[NE + e]], 1u);   // dst = edge_index[1]
}

__global__ void k_dinv(const unsigned* __restrict__ deg, float* __restrict__ dinv) {
  int n = blockIdx.x * blockDim.x + threadIdx.x;
  if (n < NN) dinv[n] = rsqrtf((float)deg[n]);
}

// exclusive scan of deg (4096) with one 1024-thread workgroup -> row_ptr, cursor
__global__ void k_scan(const unsigned* __restrict__ deg,
                       unsigned* __restrict__ rowp, unsigned* __restrict__ cur) {
  __shared__ unsigned part[1024];
  int t = threadIdx.x;
  unsigned c0 = deg[t*4+0], c1 = deg[t*4+1], c2 = deg[t*4+2], c3 = deg[t*4+3];
  unsigned lsum = c0 + c1 + c2 + c3;
  part[t] = lsum;
  __syncthreads();
  for (int off = 1; off < 1024; off <<= 1) {
    unsigned x = (t >= off) ? part[t - off] : 0u;
    __syncthreads();
    part[t] += x;
    __syncthreads();
  }
  unsigned excl = part[t] - lsum;
  unsigned r0 = excl, r1 = r0 + c0, r2 = r1 + c1, r3 = r2 + c2;
  rowp[t*4+0] = r0; rowp[t*4+1] = r1; rowp[t*4+2] = r2; rowp[t*4+3] = r3;
  cur [t*4+0] = r0; cur [t*4+1] = r1; cur [t*4+2] = r2; cur [t*4+3] = r3;
  if (t == 1023) rowp[NN] = part[1023];
}

// build CSR (col, norm) including self loops; accumulate rs[n] = sum of incoming norms
__global__ void k_scatter(const int* __restrict__ ei, const float* __restrict__ dinv,
                          unsigned* __restrict__ cur, unsigned* __restrict__ col,
                          float* __restrict__ en, float* __restrict__ rs) {
  int e = blockIdx.x * blockDim.x + threadIdx.x;
  if (e < NE) {
    int s0 = ei[e], d0 = ei[NE + e];
    float nm = dinv[s0] * dinv[d0];
    unsigned p = atomicAdd(&cur[d0], 1u);
    col[p] = (unsigned)s0; en[p] = nm;
    atomicAdd(&rs[d0], nm);
  } else if (e < NEDGE) {
    int n = e - NE;
    float dv = dinv[n], nm = dv * dv;
    unsigned p = atomicAdd(&cur[n], 1u);
    col[p] = (unsigned)n; en[p] = nm;
    atomicAdd(&rs[n], nm);
  }
}

// c0 = A_w[:, :8192] @ z0   (one wave per output row, streamed non-temporally)
__global__ void k_c0(const float* __restrict__ A_w, const float* __restrict__ z,
                     float* __restrict__ c0) {
  int tid = blockIdx.x * blockDim.x + threadIdx.x;
  int w = tid >> 5, lane = tid & 31;
  if (w >= VD) return;
  const float* row = A_w + (size_t)w * 8200;
  float s = 0.0f;
  for (int k = lane; k < VD; k += 32)
    s += __builtin_nontemporal_load(row + k) * z[k];
  for (int o = 16; o > 0; o >>= 1) s += __shfl_xor(s, o);
  if (lane == 0) c0[w] = s;
}

// d[t][m] = c0[m] + sum_j s_t[j]*(A_w[m,8192+j] + B_w[m,8192+j] + b_w[m,j])
__global__ void k_dsteps(const float* __restrict__ c0, const float* __restrict__ A_w,
                         const float* __restrict__ B_w, const float* __restrict__ b_w,
                         const float* __restrict__ semb, float* __restrict__ d) {
  int i = blockIdx.x * blockDim.x + threadIdx.x;
  if (i >= NSTEPS * VD) return;
  int t = i >> 13, m = i & (VD - 1);
  float acc = c0[m];
  #pragma unroll
  for (int j = 0; j < 8; ++j) {
    float sj = semb[t * 8 + j];
    acc += sj * (A_w[(size_t)m * 8200 + VD + j] +
                 B_w[(size_t)m * 8200 + VD + j] +
                 b_w[m * 8 + j]);
  }
  d[i] = acc;
}

// Wb[r][k] = bf16(B_w[r, k])  for k<8192 : the L2-resident hot weight matrix
__global__ void k_bf16(const float* __restrict__ B_w, uint16_t* __restrict__ Wb) {
  size_t i = (size_t)blockIdx.x * blockDim.x + threadIdx.x;
  if (i >= (size_t)VD * VD) return;
  size_t r = i >> 13, k = i & (VD - 1);
  Wb[i] = f2bf(__builtin_nontemporal_load(B_w + r * 8200 + k));
}

// ---------------- per-step GNN: 8 MixHop layers, 2 channels, persistent kernel ----------------
__global__ void k_gnn(const float* __restrict__ vcur, float* __restrict__ vnext,
                      const float* __restrict__ dsteps, int step,
                      const unsigned* __restrict__ rowp, const unsigned* __restrict__ col,
                      const float* __restrict__ en, const float* __restrict__ rs,
                      const float* __restrict__ gW0, const float* __restrict__ gb0,
                      const float* __restrict__ gWh, const float* __restrict__ gbh,
                      const float* __restrict__ gsemb,
                      float* __restrict__ Xa, float* __restrict__ Xb,
                      uint16_t* __restrict__ ubf, unsigned* __restrict__ bar) {
  const int n = blockIdx.x * blockDim.x + threadIdx.x;   // one node per thread
  float s[8];
  #pragma unroll
  for (int j = 0; j < 8; ++j) s[j] = gsemb[step * 8 + j];

  // layer-0 folded constants (step-emb channels collapse: prop(bcast s)=rs[n]*s)
  float w00[2][2], w01[2][2], k0[2], g0[2];
  #pragma unroll
  for (int o = 0; o < 2; ++o) {
    w00[0][o] = gW0[0 * 2 + o];       w00[1][o] = gW0[1 * 2 + o];
    w01[0][o] = gW0[20 + 0 * 2 + o];  w01[1][o] = gW0[20 + 1 * 2 + o];
    float t0 = gb0[o] + gb0[2 + o], t1 = 0.0f;
    #pragma unroll
    for (int j = 0; j < 8; ++j) {
      t0 += s[j] * gW0[(2 + j) * 2 + o];
      t1 += s[j] * gW0[20 + (2 + j) * 2 + o];
    }
    k0[o] = t0; g0[o] = t1;
  }

  // prefill vnext = d[step] (GEMV kernel atomically adds Wb@u on top)
  vnext[n]      = dsteps[(size_t)step * VD + n];
  vnext[NN + n] = dsteps[(size_t)step * VD + NN + n];

  unsigned p0 = rowp[n], p1 = rowp[n + 1];

  // layer 0: gather directly from v (X0 is a strided view of v)
  float px = 0.0f, py = 0.0f;
  for (unsigned p = p0; p < p1; ++p) {
    unsigned si = col[p]; float nm = en[p];
    px += nm * vcur[si]; py += nm * vcur[NN + si];
  }
  float ax = vcur[n], ay = vcur[NN + n], r = rs[n];
  float x0 = fmaxf(0.0f, ax*w00[0][0] + ay*w00[1][0] + px*w01[0][0] + py*w01[1][0] + r*g0[0] + k0[0]);
  float x1 = fmaxf(0.0f, ax*w00[0][1] + ay*w00[1][1] + px*w01[0][1] + py*w01[1][1] + r*g0[1] + k0[1]);
  ((float2*)Xa)[n] = make_float2(x0, x1);
  gbar(bar);

  const float2* Xp = (const float2*)Xa;
  float2*       Xn = (float2*)Xb;
  for (int l = 1; l < 8; ++l) {
    const float* W0 = gWh + (size_t)((l - 1) * 2 + 0) * 4;   // [i][o] 2x2
    const float* W1 = gWh + (size_t)((l - 1) * 2 + 1) * 4;
    float bs0 = gbh[((l - 1) * 2 + 0) * 2 + 0] + gbh[((l - 1) * 2 + 1) * 2 + 0];
    float bs1 = gbh[((l - 1) * 2 + 0) * 2 + 1] + gbh[((l - 1) * 2 + 1) * 2 + 1];
    float qx = 0.0f, qy = 0.0f;
    for (unsigned p = p0; p < p1; ++p) {
      unsigned si = col[p]; float nm = en[p];
      float2 xv = Xp[si];
      qx += nm * xv.x; qy += nm * xv.y;
    }
    float2 a = Xp[n];
    float o0 = fmaxf(0.0f, a.x*W0[0] + a.y*W0[2] + qx*W1[0] + qy*W1[2] + bs0);
    float o1 = fmaxf(0.0f, a.x*W0[1] + a.y*W0[3] + qx*W1[1] + qy*W1[3] + bs1);
    if (l < 7) {
      Xn[n] = make_float2(o0, o1);
      const float2* t2 = Xp; Xp = (const float2*)Xn; Xn = (float2*)t2;
      gbar(bar);
    } else {
      // u = X.T.flatten : u[c*4096+n] = X[n][c], emitted in bf16 for the WMMA GEMV
      ubf[n]      = f2bf(o0);
      ubf[NN + n] = f2bf(o1);
    }
  }
}

// ---------------- per-step GEMV: vnext += Wb(bf16, L2-resident) @ u via WMMA ----------------
// 512 row-tiles x KSPLIT K-partitions = 4096 waves; 32 WMMAs each, 4-deep pipelined loads
__global__ void k_gemv(const uint16_t* __restrict__ Wb, const uint16_t* __restrict__ ubf,
                       float* __restrict__ vnext) {
  int tid  = blockIdx.x * blockDim.x + threadIdx.x;
  int w    = tid >> 5, lane = tid & 31;
  int rt   = w & 511;        // row tile (16 rows)
  int ks   = w >> 9;         // K split (1024 elements each)
  const uint16_t* Arow = Wb + (size_t)(rt * 16 + (lane & 15)) * VD;
  const int cofs = (lane & 16) ? 8 : 0;    // A: K 0-7/16-23 vs 8-15/24-31 per lane half
  const int bofs = (lane & 16) ? 8 : 0;    // B: K 0-7/16-23 vs 8-15/24-31 per lane half
  v8f acc = {};
  const int NT = (VD / 32) / KSPLIT;       // 32 k-tiles per wave
  int kt = ks * NT;
  for (int it = 0; it < NT / 4; ++it, kt += 4) {
    // issue all 16 loads for 4 k-tiles up front; waits interleave with WMMAs
    const uint16_t* a = Arow + kt * 32 + cofs;
    const uint16_t* b = ubf  + kt * 32 + bofs;
    BfPack A0, B0, A1, B1, A2, B2, A3, B3;
    A0.s.lo = *(const v4u*)(a +  0); A0.s.hi = *(const v4u*)(a + 16);
    A1.s.lo = *(const v4u*)(a + 32); A1.s.hi = *(const v4u*)(a + 48);
    A2.s.lo = *(const v4u*)(a + 64); A2.s.hi = *(const v4u*)(a + 80);
    A3.s.lo = *(const v4u*)(a + 96); A3.s.hi = *(const v4u*)(a + 112);
    B0.s.lo = *(const v4u*)(b +  0); B0.s.hi = *(const v4u*)(b + 16);
    B1.s.lo = *(const v4u*)(b + 32); B1.s.hi = *(const v4u*)(b + 48);
    B2.s.lo = *(const v4u*)(b + 64); B2.s.hi = *(const v4u*)(b + 80);
    B3.s.lo = *(const v4u*)(b + 96); B3.s.hi = *(const v4u*)(b + 112);
    acc = __builtin_amdgcn_wmma_f32_16x16x32_bf16(false, A0.v, false, B0.v, (short)0, acc, false, false);
    acc = __builtin_amdgcn_wmma_f32_16x16x32_bf16(false, A1.v, false, B1.v, (short)0, acc, false, false);
    acc = __builtin_amdgcn_wmma_f32_16x16x32_bf16(false, A2.v, false, B2.v, (short)0, acc, false, false);
    acc = __builtin_amdgcn_wmma_f32_16x16x32_bf16(false, A3.v, false, B3.v, (short)0, acc, false, false);
  }
  // all D columns identical; column 0 lives in lane 0 (rows 0-7) and lane 16 (rows 8-15)
  if ((lane & 15) == 0) {
    int rbase = rt * 16 + ((lane >> 4) << 3);
    #pragma unroll
    for (int r0 = 0; r0 < 8; ++r0) atomicAdd(&vnext[rbase + r0], acc[r0]);
  }
}

__global__ void k_copy(const float* __restrict__ src, float* __restrict__ dst) {
  int i = blockIdx.x * blockDim.x + threadIdx.x;
  if (i < VD) dst[i] = src[i];
}

// ---------------- host ----------------
extern "C" void kernel_launch(void* const* d_in, const int* in_sizes, int n_in,
                              void* d_out, int out_size, void* d_ws, size_t ws_size,
                              hipStream_t stream) {
  (void)in_sizes; (void)n_in; (void)out_size; (void)ws_size;
  const float* z    = (const float*)d_in[0];
  const int*   ei   = (const int*)  d_in[1];
  const float* A_w  = (const float*)d_in[2];
  const float* B_w  = (const float*)d_in[3];
  const float* b_w  = (const float*)d_in[4];
  const float* semb = (const float*)d_in[5];
  const float* gsem = (const float*)d_in[6];
  const float* gW0  = (const float*)d_in[7];
  const float* gb0  = (const float*)d_in[8];
  const float* gWh  = (const float*)d_in[9];
  const float* gbh  = (const float*)d_in[10];

  char* ws = (char*)d_ws;
  size_t off = 0;
  auto alloc = [&](size_t bytes) { size_t o = off; off = (off + bytes + 255) & ~(size_t)255; return o; };
  uint16_t* Wb    = (uint16_t*)(ws + alloc((size_t)VD * VD * 2));     // 134 MB bf16 weights
  float*    dstep = (float*)   (ws + alloc((size_t)NSTEPS * VD * 4)); // d[t]
  float*    c0    = (float*)   (ws + alloc(VD * 4));
  float*    va    = (float*)   (ws + alloc(VD * 4));
  float*    vb    = (float*)   (ws + alloc(VD * 4));
  uint16_t* ubf   = (uint16_t*)(ws + alloc(VD * 2));
  float*    Xa    = (float*)   (ws + alloc(NN * 2 * 4));
  float*    Xb    = (float*)   (ws + alloc(NN * 2 * 4));
  unsigned* deg   = (unsigned*)(ws + alloc(NN * 4));
  float*    dinv  = (float*)   (ws + alloc(NN * 4));
  float*    rs    = (float*)   (ws + alloc(NN * 4));
  unsigned* rowp  = (unsigned*)(ws + alloc((NN + 1) * 4));
  unsigned* cur   = (unsigned*)(ws + alloc(NN * 4));
  unsigned* col   = (unsigned*)(ws + alloc(NEDGE * 4));
  float*    en    = (float*)   (ws + alloc(NEDGE * 4));
  unsigned* bar   = (unsigned*)(ws + alloc(256));

  // ---- one-time precompute ----
  hipLaunchKernelGGL(k_init,    dim3(32),     dim3(256), 0, stream, va, rs, deg, bar);
  hipLaunchKernelGGL(k_count,   dim3(256),    dim3(256), 0, stream, ei, deg);
  hipLaunchKernelGGL(k_dinv,    dim3(16),     dim3(256), 0, stream, deg, dinv);
  hipLaunchKernelGGL(k_scan,    dim3(1),      dim3(1024),0, stream, deg, rowp, cur);
  hipLaunchKernelGGL(k_scatter, dim3(272),    dim3(256), 0, stream, ei, dinv, cur, col, en, rs);
  hipLaunchKernelGGL(k_c0,      dim3(1024),   dim3(256), 0, stream, A_w, z, c0);
  hipLaunchKernelGGL(k_dsteps,  dim3(3200),   dim3(256), 0, stream, c0, A_w, B_w, b_w, semb, dstep);
  hipLaunchKernelGGL(k_bf16,    dim3(262144), dim3(256), 0, stream, B_w, Wb);

  // ---- 100 sequential steps: GNN (16 WGs, internal device barrier) + WMMA GEMV ----
  for (int t = 0; t < NSTEPS; ++t) {
    const float* vc = (t & 1) ? vb : va;
    float*       vn = (t & 1) ? va : vb;
    hipLaunchKernelGGL(k_gnn,  dim3(GNN_WGS), dim3(256), 0, stream,
                       vc, vn, dstep, t, rowp, col, en, rs,
                       gW0, gb0, gWh, gbh, gsem, Xa, Xb, ubf, bar);
    hipLaunchKernelGGL(k_gemv, dim3(512),     dim3(256), 0, stream, Wb, ubf, vn);
  }
  // t=99 (odd) wrote va
  hipLaunchKernelGGL(k_copy, dim3(32), dim3(256), 0, stream, va, (float*)d_out);
}
